// GNNNaiveBlock_ChebGat_3435973837208
// MI455X (gfx1250) — matrix operand order repaired and checked by reading, not compiled, not run on hardware
//
#include <hip/hip_runtime.h>
#include <hip/hip_bf16.h>
#include <math.h>

typedef __attribute__((ext_vector_type(16))) _Float16 v16h;
typedef __attribute__((ext_vector_type(8)))  _Float16 v8h;
typedef __attribute__((ext_vector_type(8)))  float    v8f;

#define LN_EPS   1e-5f
#define SM_EPS   1e-16f
#define NEG_ACT  0.01f
#define NEG_GAT  0.2f
#define HEADS    4
#define FDIM     64

// ---------------------------------------------------------------- utilities

__global__ void fill_kernel(float* __restrict__ p, size_t n, float v) {
    size_t i = (size_t)blockIdx.x * blockDim.x + threadIdx.x;
    if (i < n) p[i] = v;
}

__global__ void cvt_f16_kernel(const float* __restrict__ in, _Float16* __restrict__ out, size_t n) {
    size_t i = (size_t)blockIdx.x * blockDim.x + threadIdx.x;
    if (i < n) out[i] = (_Float16)in[i];
}

__device__ __forceinline__ float leaky(float x, float s) {
    return x >= 0.f ? x : s * x;
}

__device__ __forceinline__ void atomicMaxF(float* addr, float val) {
    int* ai = (int*)addr;
    int old = __float_as_int(*addr);
    while (__int_as_float(old) < val) {
        int assumed = old;
        old = atomicCAS(ai, assumed, __float_as_int(val));
        if (old == assumed) break;
    }
}

// -------------------------------------------------- weight fragment packing
// B fragment layout (32x16 f16 -> 32 lanes x 16 halves):
//   lane l covers col = l&15 ; halves j cover K = kb + (l>=16?16:0) + j
// Packed as P[frag*512 + lane*16 + j] so a wave loads one fragment with a
// single contiguous 32B-per-lane load.

__global__ void pack_chebw_kernel(const float* __restrict__ W, _Float16* __restrict__ P) {
    int t = blockIdx.x * blockDim.x + threadIdx.x;          // 3*2*4*512 = 12288
    if (t >= 3 * 2 * 4 * 512) return;
    int j    = t & 15;
    int lane = (t >> 4) & 31;
    int frag = t >> 9;            // (m*2+kc)*4 + ct
    int ct   = frag & 3;
    int kc   = (frag >> 2) & 1;
    int m    = frag >> 3;
    int k    = kc * 32 + ((lane >> 4) << 4) + j;
    int col  = ct * 16 + (lane & 15);
    P[t] = (_Float16)W[(m * FDIM + k) * FDIM + col];
}

__global__ void pack_gatw_kernel(const float* __restrict__ W, _Float16* __restrict__ P) {
    int t = blockIdx.x * blockDim.x + threadIdx.x;          // 2*16*512 = 16384
    if (t >= 2 * 16 * 512) return;
    int j    = t & 15;
    int lane = (t >> 4) & 31;
    int frag = t >> 9;            // kc*16 + ct
    int ct   = frag & 15;
    int kc   = frag >> 4;
    int k    = kc * 32 + ((lane >> 4) << 4) + j;
    int col  = ct * 16 + (lane & 15);
    P[t] = (_Float16)W[k * (HEADS * FDIM) + col];
}

// ---------------------------------------------------------------- layernorm

__global__ void layernorm_kernel(const float* __restrict__ x,
                                 const float* __restrict__ gamma,
                                 const float* __restrict__ beta,
                                 float* __restrict__ xn,
                                 _Float16* __restrict__ xn16, int n) {
    int r = blockIdx.x * blockDim.x + threadIdx.x;
    if (r >= n) return;
    const float* xr = x + (size_t)r * FDIM;
    float mu = 0.f;
    #pragma unroll 8
    for (int c = 0; c < FDIM; ++c) mu += xr[c];
    mu *= (1.f / FDIM);
    float var = 0.f;
    #pragma unroll 8
    for (int c = 0; c < FDIM; ++c) { float d = xr[c] - mu; var += d * d; }
    var *= (1.f / FDIM);
    float rstd = rsqrtf(var + LN_EPS);
    float* o = xn + (size_t)r * FDIM;
    _Float16* o16 = xn16 + (size_t)r * FDIM;
    #pragma unroll 8
    for (int c = 0; c < FDIM; ++c) {
        float v = (xr[c] - mu) * rstd * gamma[c] + beta[c];
        o[c] = v;
        o16[c] = (_Float16)v;
    }
}

// ---------------------------------------------------------------- graph prep

__global__ void degree_kernel(const int* __restrict__ src, const float* __restrict__ ea,
                              float* __restrict__ deg, int E) {
    int e = blockIdx.x * blockDim.x + threadIdx.x;
    if (e < E) atomicAdd(&deg[src[e]], ea[e]);
}

__global__ void dinv_kernel(const float* __restrict__ deg, float* __restrict__ dinv, int n) {
    int i = blockIdx.x * blockDim.x + threadIdx.x;
    if (i < n) {
        float d = deg[i];
        dinv[i] = d > 0.f ? rsqrtf(d) : 0.f;
    }
}

__global__ void edgew_kernel(const int* __restrict__ src, const int* __restrict__ dst,
                             const float* __restrict__ ea, const float* __restrict__ dinv,
                             float* __restrict__ w, int E) {
    int e = blockIdx.x * blockDim.x + threadIdx.x;
    if (e < E) w[e] = -(dinv[src[e]] * ea[e] * dinv[dst[e]]);
}

// xout[dst] += w[e] * xin[src]   (one wave per edge, float2 per lane)
__global__ void prop_kernel(const float* __restrict__ xin, float* __restrict__ xout,
                            const int* __restrict__ src, const int* __restrict__ dst,
                            const float* __restrict__ w, int E) {
    size_t t = (size_t)blockIdx.x * blockDim.x + threadIdx.x;
    int e = (int)(t >> 5);
    int lane = (int)(t & 31);
    if (e >= E) return;
    int s = src[e], d = dst[e];
    float wv = w[e];
    float2 v = *(const float2*)(xin + (size_t)s * FDIM + lane * 2);
    atomicAdd(xout + (size_t)d * FDIM + lane * 2,     wv * v.x);
    atomicAdd(xout + (size_t)d * FDIM + lane * 2 + 1, wv * v.y);
}

// Tx2 = 2*P - Tx0 (in place) and emit f16 copy
__global__ void cheb_combine_kernel(float* __restrict__ P, const float* __restrict__ Tx0,
                                    _Float16* __restrict__ P16, size_t n) {
    size_t i = (size_t)blockIdx.x * blockDim.x + threadIdx.x;
    if (i < n) {
        float v = 2.f * P[i] - Tx0[i];
        P[i] = v;
        P16[i] = (_Float16)v;
    }
}

// ---------------------------------------------------------------- WMMA GEMMs
// A fragment (16x32 f16, contiguous k order, as packed B): lane l, row = l&15;
//   halves 0..7  : K = kb + (l>=16? 16:0) + j
//   halves 8..15 : K = kb + 8 + (l>=16? 16:0) + j
// D fragment (16x16 f32): lane l, col = l&15; VGPR i: row = i + (l>=16? 8:0)

__device__ __forceinline__ v16h load_a_frag(const _Float16* __restrict__ rowptr, int kb, int lane) {
    int koff = kb + ((lane >> 4) << 4);
    v8h lo = *(const v8h*)(rowptr + koff);
    v8h hi = *(const v8h*)(rowptr + koff + 8);
    v16h a;
    #pragma unroll
    for (int j = 0; j < 8; ++j) { a[j] = lo[j]; a[j + 8] = hi[j]; }
    return a;
}

// h16 = f16(leaky(Tx0@W0 + Tx1@W1 + Tx2@W2 + bias))
__global__ __launch_bounds__(128) void cheb_gemm_kernel(
        const _Float16* __restrict__ Tx0, const _Float16* __restrict__ Tx1,
        const _Float16* __restrict__ Tx2,
        const _Float16* __restrict__ Wp,  // packed fragments [(m*2+kc)*4+ct][512]
        const float* __restrict__ bias,
        _Float16* __restrict__ h16, int n) {
    int lane = threadIdx.x & 31;
    int wave = threadIdx.x >> 5;       // column tile 0..3
    int row0 = blockIdx.x * 16;
    int lrow = lane & 15;
    int arow = row0 + lrow;  if (arow >= n) arow = n - 1;
    v8f acc = {};
    const _Float16* mats[3] = { Tx0, Tx1, Tx2 };
    #pragma unroll
    for (int m = 0; m < 3; ++m) {
        const _Float16* A = mats[m] + (size_t)arow * FDIM;
        #pragma unroll
        for (int kc = 0; kc < 2; ++kc) {
            v16h a = load_a_frag(A, kc * 32, lane);
            v16h b = *(const v16h*)(Wp + (size_t)(((m * 2 + kc) * 4 + wave) * 512 + lane * 16));
            acc = __builtin_amdgcn_wmma_f32_16x16x32_f16(false, a, false, b,
                                                         (short)0, acc, false, false);
        }
    }
    int colOut = wave * 16 + lrow;
    float bv = bias[colOut];
    int rbase = row0 + ((lane >> 4) << 3);
    #pragma unroll
    for (int i = 0; i < 8; ++i) {
        int r = rbase + i;
        if (r < n) h16[(size_t)r * FDIM + colOut] = (_Float16)leaky(acc[i] + bv, NEG_ACT);
    }
}

// xh[N,256] = h @ gat_w
__global__ __launch_bounds__(128) void gat_gemm_kernel(
        const _Float16* __restrict__ h16, const _Float16* __restrict__ Wp, // [kc*16+ct][512]
        float* __restrict__ xh, int n) {
    int lane = threadIdx.x & 31;
    int wave = threadIdx.x >> 5;
    int row0 = blockIdx.x * 16;
    int ct = blockIdx.y * 4 + wave;    // column tile 0..15
    int lrow = lane & 15;
    int arow = row0 + lrow;  if (arow >= n) arow = n - 1;
    const _Float16* A = h16 + (size_t)arow * FDIM;
    v8f acc = {};
    #pragma unroll
    for (int kc = 0; kc < 2; ++kc) {
        v16h a = load_a_frag(A, kc * 32, lane);
        v16h b = *(const v16h*)(Wp + (size_t)((kc * 16 + ct) * 512 + lane * 16));
        acc = __builtin_amdgcn_wmma_f32_16x16x32_f16(false, a, false, b,
                                                     (short)0, acc, false, false);
    }
    int colOut = ct * 16 + lrow;
    int rbase = row0 + ((lane >> 4) << 3);
    #pragma unroll
    for (int i = 0; i < 8; ++i) {
        int r = rbase + i;
        if (r < n) xh[(size_t)r * (HEADS * FDIM) + colOut] = acc[i];
    }
}

// ---------------------------------------------------------------- GAT attention

__global__ void attn_scores_kernel(const float* __restrict__ xh,
                                   const float* __restrict__ att_src,
                                   const float* __restrict__ att_dst,
                                   float* __restrict__ a_src, float* __restrict__ a_dst, int n) {
    int t = blockIdx.x * blockDim.x + threadIdx.x;
    if (t >= n * HEADS) return;
    int node = t >> 2, hh = t & 3;
    const float* xp = xh + (size_t)node * (HEADS * FDIM) + hh * FDIM;
    const float* ws = att_src + hh * FDIM;
    const float* wd = att_dst + hh * FDIM;
    float as = 0.f, ad = 0.f;
    #pragma unroll 8
    for (int c = 0; c < FDIM; ++c) { float v = xp[c]; as += v * ws[c]; ad += v * wd[c]; }
    a_src[t] = as; a_dst[t] = ad;
}

__device__ __forceinline__ void edge_sd(int e, int E, const int* src, const int* dst,
                                        int* s, int* d) {
    if (e < E) { *s = src[e]; *d = dst[e]; } else { *s = *d = e - E; }
}

__global__ void edge_max_kernel(const float* __restrict__ a_src, const float* __restrict__ a_dst,
                                const int* __restrict__ src, const int* __restrict__ dst,
                                float* __restrict__ emax, int E, int n) {
    int t = blockIdx.x * blockDim.x + threadIdx.x;
    int total = (E + n) * HEADS;
    if (t >= total) return;
    int e = t >> 2, hh = t & 3;
    int s, d; edge_sd(e, E, src, dst, &s, &d);
    float ev = leaky(a_src[s * HEADS + hh] + a_dst[d * HEADS + hh], NEG_GAT);
    atomicMaxF(&emax[d * HEADS + hh], ev);
}

__global__ void edge_expsum_kernel(const float* __restrict__ a_src, const float* __restrict__ a_dst,
                                   const int* __restrict__ src, const int* __restrict__ dst,
                                   const float* __restrict__ emax, float* __restrict__ denom,
                                   int E, int n) {
    int t = blockIdx.x * blockDim.x + threadIdx.x;
    int total = (E + n) * HEADS;
    if (t >= total) return;
    int e = t >> 2, hh = t & 3;
    int s, d; edge_sd(e, E, src, dst, &s, &d);
    float ev = leaky(a_src[s * HEADS + hh] + a_dst[d * HEADS + hh], NEG_GAT);
    float ex = expf(ev - emax[d * HEADS + hh]);
    atomicAdd(&denom[d * HEADS + hh], ex);
}

// one wave per edge slot; head-mean folded into accumulation -> o is [N,64]
__global__ void gat_aggregate_kernel(const float* __restrict__ xh,
                                     const float* __restrict__ a_src, const float* __restrict__ a_dst,
                                     const float* __restrict__ emax, const float* __restrict__ denom,
                                     const int* __restrict__ src, const int* __restrict__ dst,
                                     float* __restrict__ o, int E, int n) {
    size_t t = (size_t)blockIdx.x * blockDim.x + threadIdx.x;
    int e = (int)(t >> 5);
    int lane = (int)(t & 31);
    if (e >= E + n) return;
    int s, d; edge_sd(e, E, src, dst, &s, &d);
    float acc0 = 0.f, acc1 = 0.f;
    #pragma unroll
    for (int hh = 0; hh < HEADS; ++hh) {
        float ev = leaky(a_src[s * HEADS + hh] + a_dst[d * HEADS + hh], NEG_GAT);
        float ex = expf(ev - emax[d * HEADS + hh]);
        float alpha = ex / (denom[d * HEADS + hh] + SM_EPS);
        float2 xv = *(const float2*)(xh + (size_t)s * (HEADS * FDIM) + hh * FDIM + lane * 2);
        acc0 += alpha * xv.x;
        acc1 += alpha * xv.y;
    }
    atomicAdd(o + (size_t)d * FDIM + lane * 2,     acc0 * (1.f / HEADS));
    atomicAdd(o + (size_t)d * FDIM + lane * 2 + 1, acc1 * (1.f / HEADS));
}

__global__ void finalize_kernel(const float* __restrict__ o, const float* __restrict__ bias,
                                float* __restrict__ out, size_t total) {
    size_t i = (size_t)blockIdx.x * blockDim.x + threadIdx.x;
    if (i >= total) return;
    int c = (int)(i & (FDIM - 1));
    out[i] = leaky(o[i] + bias[c], NEG_ACT);
}

// ---------------------------------------------------------------- launcher

extern "C" void kernel_launch(void* const* d_in, const int* in_sizes, int n_in,
                              void* d_out, int out_size, void* d_ws, size_t ws_size,
                              hipStream_t stream) {
    const float* x        = (const float*)d_in[0];
    const int*   eidx     = (const int*)d_in[1];
    const float* eattr    = (const float*)d_in[2];
    const float* ln_gamma = (const float*)d_in[3];
    const float* ln_beta  = (const float*)d_in[4];
    const float* cheb_w   = (const float*)d_in[5];
    const float* cheb_b   = (const float*)d_in[6];
    const float* gat_w    = (const float*)d_in[7];
    const float* att_src  = (const float*)d_in[8];
    const float* att_dst  = (const float*)d_in[9];
    const float* gat_b    = (const float*)d_in[10];

    const int N = in_sizes[0] / FDIM;
    const int E = in_sizes[1] / 2;
    const int* src = eidx;
    const int* dst = eidx + E;

    // ---- workspace layout (f32 units) ----
    float* wsf = (float*)d_ws;
    size_t off = 0;
    float* xn    = wsf + off; off += (size_t)N * FDIM;           // Tx0 (f32)
    float* Tx1   = wsf + off; off += (size_t)N * FDIM;
    float* Tx2   = wsf + off; off += (size_t)N * FDIM;           // also P scratch
    float* xh    = wsf + off; off += (size_t)N * HEADS * FDIM;
    float* obuf  = wsf + off; off += (size_t)N * FDIM;
    float* deg   = wsf + off; off += (size_t)N;
    float* dinv  = wsf + off; off += (size_t)N;
    float* wbuf  = wsf + off; off += (size_t)E;
    float* a_src = wsf + off; off += (size_t)N * HEADS;
    float* a_dst = wsf + off; off += (size_t)N * HEADS;
    float* emax  = wsf + off; off += (size_t)N * HEADS;
    float* denom = wsf + off; off += (size_t)N * HEADS;
    off = (off + 7) & ~(size_t)7;   // 32B align for f16 vector loads
    _Float16* xn16  = (_Float16*)(wsf + off); off += (size_t)N * FDIM / 2;
    _Float16* Tx116 = (_Float16*)(wsf + off); off += (size_t)N * FDIM / 2;
    _Float16* Tx216 = (_Float16*)(wsf + off); off += (size_t)N * FDIM / 2;
    _Float16* h16   = (_Float16*)(wsf + off); off += (size_t)N * FDIM / 2;
    _Float16* chebP = (_Float16*)(wsf + off); off += (3 * 2 * 4 * 512) / 2;
    _Float16* gatP  = (_Float16*)(wsf + off); off += (2 * 16 * 512) / 2;

    const int T = 256;
    auto gr = [](size_t n, int t) { return (unsigned)((n + t - 1) / t); };

    // pack weights into WMMA B-fragment layout
    pack_chebw_kernel<<<gr(3 * 2 * 4 * 512, T), T, 0, stream>>>(cheb_w, chebP);
    pack_gatw_kernel<<<gr(2 * 16 * 512, T), T, 0, stream>>>(gat_w, gatP);

    // layernorm (emits f32 + f16 copies)
    layernorm_kernel<<<gr(N, T), T, 0, stream>>>(x, ln_gamma, ln_beta, xn, xn16, N);

    // scaled Laplacian edge weights
    fill_kernel<<<gr(N, T), T, 0, stream>>>(deg, (size_t)N, 0.f);
    degree_kernel<<<gr(E, T), T, 0, stream>>>(src, eattr, deg, E);
    dinv_kernel<<<gr(N, T), T, 0, stream>>>(deg, dinv, N);
    edgew_kernel<<<gr(E, T), T, 0, stream>>>(src, dst, eattr, dinv, wbuf, E);

    // Tx1 = L_hat @ Tx0
    fill_kernel<<<gr((size_t)N * FDIM, T), T, 0, stream>>>(Tx1, (size_t)N * FDIM, 0.f);
    prop_kernel<<<gr((size_t)E * 32, T), T, 0, stream>>>(xn, Tx1, src, dst, wbuf, E);
    cvt_f16_kernel<<<gr((size_t)N * FDIM, T), T, 0, stream>>>(Tx1, Tx116, (size_t)N * FDIM);
    // Tx2 = 2 * (L_hat @ Tx1) - Tx0
    fill_kernel<<<gr((size_t)N * FDIM, T), T, 0, stream>>>(Tx2, (size_t)N * FDIM, 0.f);
    prop_kernel<<<gr((size_t)E * 32, T), T, 0, stream>>>(Tx1, Tx2, src, dst, wbuf, E);
    cheb_combine_kernel<<<gr((size_t)N * FDIM, T), T, 0, stream>>>(Tx2, xn, Tx216,
                                                                   (size_t)N * FDIM);

    // h16 = leaky(sum_k Txk @ Wk + b)   (WMMA, f16 output)
    unsigned rtiles = (unsigned)((N + 15) / 16);
    cheb_gemm_kernel<<<rtiles, 128, 0, stream>>>(xn16, Tx116, Tx216, chebP, cheb_b, h16, N);

    // xh = h @ gat_w   (WMMA)
    gat_gemm_kernel<<<dim3(rtiles, 4), 128, 0, stream>>>(h16, gatP, xh, N);

    // attention
    attn_scores_kernel<<<gr((size_t)N * HEADS, T), T, 0, stream>>>(xh, att_src, att_dst,
                                                                   a_src, a_dst, N);
    fill_kernel<<<gr((size_t)N * HEADS, T), T, 0, stream>>>(emax, (size_t)N * HEADS, -3.0e38f);
    fill_kernel<<<gr((size_t)N * HEADS, T), T, 0, stream>>>(denom, (size_t)N * HEADS, 0.f);
    edge_max_kernel<<<gr((size_t)(E + N) * HEADS, T), T, 0, stream>>>(a_src, a_dst, src, dst,
                                                                      emax, E, N);
    edge_expsum_kernel<<<gr((size_t)(E + N) * HEADS, T), T, 0, stream>>>(a_src, a_dst, src, dst,
                                                                         emax, denom, E, N);
    fill_kernel<<<gr((size_t)N * FDIM, T), T, 0, stream>>>(obuf, (size_t)N * FDIM, 0.f);
    gat_aggregate_kernel<<<gr((size_t)(E + N) * 32, T), T, 0, stream>>>(xh, a_src, a_dst,
                                                                        emax, denom, src, dst,
                                                                        obuf, E, N);
    finalize_kernel<<<gr((size_t)N * FDIM, T), T, 0, stream>>>(obuf, gat_b, (float*)d_out,
                                                               (size_t)N * FDIM);
}
